// Module1GRU_4964982194520
// MI455X (gfx1250) — matrix operand order repaired and checked by reading, not compiled, not run on hardware
//
#include <hip/hip_runtime.h>
#include <stdint.h>

// Problem constants (from reference)
#define S_LEN  1024
#define HDIM   512
#define BATCH  8
#define NSPK   8
#define NL     2
#define N3H    1536       // 3*H
#define NTILES 96         // N3H / 16
#define KTILES 16         // HDIM / 32

typedef __attribute__((ext_vector_type(16))) __bf16 v16bf;
typedef __attribute__((ext_vector_type(4)))  __bf16 v4bf;
typedef __attribute__((ext_vector_type(8)))  float  v8f;
typedef __attribute__((ext_vector_type(4)))  float  v4f;

// ---------------------------------------------------------------------------
// Weight fragment packing (B-matrix operand, 32x16 bf16).
// fid = (((sp*2 + l)*2 + which)*NTILES + nt)*KTILES + kt ; which: 0=Wi, 1=Wh
// Per-lane layout (ISA 7.12.2 dense 16-bit B): lane L -> col N = L&15,
// K-local = (L>=16 ? 8 : 0) + elem (elems 8..15 -> +8 more, i.e. +16 base).  32 lanes x 32B contiguous per fragment.
// ---------------------------------------------------------------------------
__global__ __launch_bounds__(256) void pack_weights_kernel(
    const float* __restrict__ Wi, const float* __restrict__ Wh,
    void* __restrict__ wout_v) {
  v16bf* wout = (v16bf*)wout_v;
  int gtid = blockIdx.x * blockDim.x + threadIdx.x;
  int id   = gtid >> 5;          // fragment id
  int lane = gtid & 31;

  int kt    = id & (KTILES - 1);
  int nt    = (id >> 4) % NTILES;
  int rest  = id / (KTILES * NTILES);
  int which = rest & 1;
  int l     = (rest >> 1) & 1;
  int sp    = rest >> 2;

  const float* src = (which ? Wh : Wi) + (size_t)(sp * NL + l) * N3H * HDIM;
  int n  = nt * 16 + (lane & 15);
  int kb = kt * 32 + ((lane & 16) ? 8 : 0);
  const float* rp = src + (size_t)n * HDIM + kb;

  float4 f0 = *(const float4*)(rp + 0);
  float4 f1 = *(const float4*)(rp + 4);
  float4 f2 = *(const float4*)(rp + 16);
  float4 f3 = *(const float4*)(rp + 20);
  float vals[16] = {f0.x, f0.y, f0.z, f0.w, f1.x, f1.y, f1.z, f1.w,
                    f2.x, f2.y, f2.z, f2.w, f3.x, f3.y, f3.z, f3.w};
  v16bf o;
#pragma unroll
  for (int i = 0; i < 16; ++i) o[i] = (__bf16)vals[i];
  wout[(size_t)id * 32 + lane] = o;
}

// ---------------------------------------------------------------------------
// Pack x into A-fragment layout (16x32 bf16) for every timestep.
// Fragment id = t*KTILES + kt. Per-lane A layout (ISA 7.12.2 16-bit A):
// row M = L&15; elems 0..7 -> k = kt*32 + (L>=16?8:0) + {0..7};
// elems 8..15 -> +16. Rows 8..15 are zero padding.
// ---------------------------------------------------------------------------
__global__ __launch_bounds__(256) void pack_x_kernel(
    const float* __restrict__ x, void* __restrict__ xf_v) {
  v16bf* xf = (v16bf*)xf_v;
  int gtid = blockIdx.x * blockDim.x + threadIdx.x;
  int w    = gtid >> 5;         // fragment id: t*16 + kt
  int lane = gtid & 31;
  int kt   = w & (KTILES - 1);
  int t    = w >> 4;
  int row  = lane & 15;

  v16bf o;
#pragma unroll
  for (int i = 0; i < 16; ++i) o[i] = (__bf16)0.0f;
  if (row < 8) {
    const float* rp = x + ((size_t)row * S_LEN + t) * HDIM + kt * 32 +
                      ((lane & 16) ? 8 : 0);
    float4 a0 = *(const float4*)(rp + 0);
    float4 a1 = *(const float4*)(rp + 4);
    float4 b0 = *(const float4*)(rp + 16);
    float4 b1 = *(const float4*)(rp + 20);
    float vals[16] = {a0.x, a0.y, a0.z, a0.w, a1.x, a1.y, a1.z, a1.w,
                      b0.x, b0.y, b0.z, b0.w, b1.x, b1.y, b1.z, b1.w};
#pragma unroll
    for (int i = 0; i < 16; ++i) o[i] = (__bf16)vals[i];
  }
  xf[(size_t)w * 32 + lane] = o;
}

// ---------------------------------------------------------------------------
// Persistent GRU kernel: one workgroup per speaker, 1024 threads = 32 waves.
// ---------------------------------------------------------------------------
struct SMem {
  float bi[NL][N3H];            // 12 KB
  float bh[NL][N3H];            // 12 KB
  float h[NL][BATCH][HDIM];     // 32 KB  f32 hidden state (for gate math)
  v16bf hfrag[NL][KTILES][32];  // 32 KB  bf16 A-fragments of h
  v16bf xfrag[2][KTILES][32];   // 32 KB  double-buffered prepacked x_t
  float rz[BATCH][2 * HDIM];    // 32 KB  sigmoid(ir+hr), sigmoid(iz+hz)
  float inb[BATCH][HDIM];       // 16 KB  i_n + bias
  float hnb[BATCH][HDIM];       // 16 KB  h_n + bias
};                              // ~184 KB (< 320 KB WGP LDS)

__device__ __forceinline__ float sigmoidf_(float x) {
  return 1.0f / (1.0f + __expf(-x));
}
__device__ __forceinline__ float tanh_fast(float x) {
  float e = __expf(2.0f * x);
  return 1.0f - 2.0f / (e + 1.0f);
}

// CDNA5 async global->LDS copy of one timestep of prepacked x fragments
// (16 KB, one b128 per thread). Tracked by ASYNCcnt.
__device__ __forceinline__ void prefetch_xfrag(const char* __restrict__ src,
                                               void* dst, int tid) {
  const char* g = src + tid * 16;
  uint32_t lds = (uint32_t)(uintptr_t)((char*)dst + tid * 16);
  asm volatile("global_load_async_to_lds_b128 %0, %1, off"
               :: "v"(lds), "v"(g) : "memory");
}

__global__ __launch_bounds__(1024, 1) void gru_persistent_kernel(
    const char* __restrict__ xf, const int* __restrict__ spk,
    const float* __restrict__ bi_g, const float* __restrict__ bh_g,
    const void* __restrict__ wp_v, float* __restrict__ out) {
  extern __shared__ char smem_raw[];
  SMem& sm = *(SMem*)smem_raw;
  const v16bf* wp = (const v16bf*)wp_v;

  const int tid  = threadIdx.x;
  const int lane = tid & 31;
  const int wave = tid >> 5;      // 0..31
  const int sp   = blockIdx.x;    // speaker

  // ---- init: prefetch x(0) frags, stage biases, zero hidden state ----
  prefetch_xfrag(xf, &sm.xfrag[0][0][0], tid);
  {
    const float* bis = bi_g + (size_t)sp * NL * N3H;
    const float* bhs = bh_g + (size_t)sp * NL * N3H;
    float* dbi = &sm.bi[0][0];
    float* dbh = &sm.bh[0][0];
    for (int i = tid; i < NL * N3H; i += 1024) { dbi[i] = bis[i]; dbh[i] = bhs[i]; }
    float* hz = &sm.h[0][0][0];
    for (int i = tid; i < NL * BATCH * HDIM; i += 1024) hz[i] = 0.0f;
    uint32_t* hf = (uint32_t*)&sm.hfrag[0][0][0];
    for (int i = tid; i < NL * KTILES * 32 * 8; i += 1024) hf[i] = 0u;
  }
  asm volatile("s_wait_asynccnt 0" ::: "memory");
  __syncthreads();

  const size_t spbase = (size_t)sp * NL * 2 * NTILES * KTILES;  // fragments
  const int gb  = tid >> 7;            // gate phase: batch row 0..7
  const int gj0 = (tid & 127) << 2;    // gate phase: 4 columns per thread
  // Fragment slot this gate thread feeds (inverse of A layout):
  const int gkt    = gj0 >> 5;
  const int gkl    = gj0 & 31;
  const int gflane = gb + ((gkl & 8) ? 16 : 0);
  const int ge0    = (gkl & 7) + ((gkl & 16) ? 8 : 0);

  for (int t = 0; t < S_LEN; ++t) {
    const int buf = t & 1;
    if (t + 1 < S_LEN)
      prefetch_xfrag(xf + (size_t)(t + 1) * (KTILES * 1024),
                     &sm.xfrag[buf ^ 1][0][0], tid);
    const bool m = (spk[gb * S_LEN + t] == sp);

#pragma unroll
    for (int l = 0; l < NL; ++l) {
      v16bf (*Af)[32]  = (l == 0) ? sm.xfrag[buf] : sm.hfrag[0];
      v16bf (*Ahf)[32] = sm.hfrag[l];

      // ---- GEMM: wave owns 3 N-tiles; computes gi and gh for each ----
      const int nt0 = wave * 3;
      v8f acci[3], acch[3];
#pragma unroll
      for (int j = 0; j < 3; ++j)
#pragma unroll
        for (int e = 0; e < 8; ++e) { acci[j][e] = 0.0f; acch[j][e] = 0.0f; }

      const v16bf* wbi = wp + (spbase + (size_t)(l * 2 + 0) * NTILES * KTILES) * 32;
      const v16bf* wbh = wp + (spbase + (size_t)(l * 2 + 1) * NTILES * KTILES) * 32;

#pragma unroll 4
      for (int kt = 0; kt < KTILES; ++kt) {
        v16bf a_in = Af[kt][lane];    // 2x ds_load_b128, prepacked bf16
        v16bf a_h  = Ahf[kt][lane];
#pragma unroll
        for (int j = 0; j < 3; ++j) {
          int nt = nt0 + j;
          v16bf bfi = wbi[((size_t)nt * KTILES + kt) * 32 + lane];
          acci[j] = __builtin_amdgcn_wmma_f32_16x16x32_bf16(
              false, a_in, false, bfi, (short)0, acci[j], false, false);
          v16bf bfh = wbh[((size_t)nt * KTILES + kt) * 32 + lane];
          acch[j] = __builtin_amdgcn_wmma_f32_16x16x32_bf16(
              false, a_h, false, bfh, (short)0, acch[j], false, false);
        }
      }

      // Epilogue: C layout -> lanes 0..15 hold real rows M=0..7 in elems 0..7.
      if (lane < 16) {
#pragma unroll
        for (int j = 0; j < 3; ++j) {
          int nt = nt0 + j;
          int n  = nt * 16 + lane;
          float bv_i = sm.bi[l][n];
          float bv_h = sm.bh[l][n];
          if (nt < 64) {  // r and z gates: fuse sum + sigmoid in registers
#pragma unroll
            for (int r = 0; r < 8; ++r)
              sm.rz[r][n] = sigmoidf_(acci[j][r] + bv_i + acch[j][r] + bv_h);
          } else {        // n-gate halves kept separate for r * h_n
#pragma unroll
            for (int r = 0; r < 8; ++r) {
              sm.inb[r][n - 1024] = acci[j][r] + bv_i;
              sm.hnb[r][n - 1024] = acch[j][r] + bv_h;
            }
          }
        }
      }
      __syncthreads();

      // ---- Gate phase: 4096 elements, 4 consecutive columns per thread ----
      {
        float4 rq = *(const float4*)&sm.rz[gb][gj0];
        float4 zq = *(const float4*)&sm.rz[gb][HDIM + gj0];
        float4 iq = *(const float4*)&sm.inb[gb][gj0];
        float4 nq = *(const float4*)&sm.hnb[gb][gj0];
        float*  hl = &sm.h[l][gb][gj0];
        float4 ho = *(const float4*)hl;
#define GRU_GATE(c)                                          \
        {                                                    \
          float nn   = tanh_fast(iq.c + rq.c * nq.c);        \
          float hnew = (1.0f - zq.c) * nn + zq.c * ho.c;     \
          ho.c = m ? hnew : ho.c;                            \
        }
        GRU_GATE(x) GRU_GATE(y) GRU_GATE(z) GRU_GATE(w)
#undef GRU_GATE
        *(float4*)hl = ho;                       // f32 state for gate math
        // bf16 A-fragment slot for next GEMM that consumes h[l]
        v4bf pk;
        pk[0] = (__bf16)ho.x; pk[1] = (__bf16)ho.y;
        pk[2] = (__bf16)ho.z; pk[3] = (__bf16)ho.w;
        *(v4bf*)((__bf16*)&sm.hfrag[l][gkt][gflane] + ge0) = pk;
        if (l == 1 && m) {
          // write-once output: non-temporal store (native ext-vector type)
          v4f hv;
          hv.x = ho.x; hv.y = ho.y; hv.z = ho.z; hv.w = ho.w;
          v4f* op = (v4f*)(out + ((size_t)gb * S_LEN + t) * HDIM + gj0);
          __builtin_nontemporal_store(hv, op);
        }
      }
      if (l == 0) __syncthreads();
    }

    asm volatile("s_wait_asynccnt 0" ::: "memory");
    __syncthreads();
  }
}

// ---------------------------------------------------------------------------
extern "C" void kernel_launch(void* const* d_in, const int* in_sizes, int n_in,
                              void* d_out, int out_size, void* d_ws, size_t ws_size,
                              hipStream_t stream) {
  const float* x   = (const float*)d_in[0];  // (8,1024,512) f32
  const int*   spk = (const int*)d_in[1];    // (8,1024) i32
  const float* Wi  = (const float*)d_in[2];  // (8,2,1536,512) f32
  const float* Wh  = (const float*)d_in[3];  // (8,2,1536,512) f32
  const float* bi  = (const float*)d_in[4];  // (8,2,1536) f32
  const float* bh  = (const float*)d_in[5];  // (8,2,1536) f32
  float* out = (float*)d_out;                // (8,1024,512) f32

  // d_ws layout: [0,48MB) bf16 weight fragments, [48MB,64MB) bf16 x fragments.
  const int nfragw = NSPK * NL * 2 * NTILES * KTILES;      // 49152 -> 48 MB
  char* wsc = (char*)d_ws;
  void* wfrag = wsc;
  void* xfrag = wsc + (size_t)nfragw * 1024;

  pack_weights_kernel<<<(nfragw * 32) / 256, 256, 0, stream>>>(Wi, Wh, wfrag);
  pack_x_kernel<<<(S_LEN * KTILES * 32) / 256, 256, 0, stream>>>(x, xfrag);

  gru_persistent_kernel<<<NSPK, 1024, sizeof(SMem), stream>>>(
      (const char*)xfrag, spk, bi, bh, wfrag, out);
}
